// ScoreMagnitudeMetric_63453846831506
// MI455X (gfx1250) — matrix operand order repaired and checked by reading, not compiled, not run on hardware
//
#include <hip/hip_runtime.h>

// ---------------- types ----------------
typedef __attribute__((ext_vector_type(16))) __bf16 v16bf;
typedef __attribute__((ext_vector_type(8)))  __bf16 v8bf;
typedef __attribute__((ext_vector_type(8)))  float  v8f;

#define DD        1024      // D (K and N of the GEMM)
#define M_BLOCK   64        // rows per workgroup
#define LDS_PITCH 1032      // padded bf16 pitch: 4-bank shift per row -> conflict-free
#define GEMM_THREADS 128    // 4 waves, 16 rows each
#define NT_COLS   64        // N columns per wave iteration = 4 WMMA tiles sharing one A frag

__device__ __forceinline__ unsigned short f2bf(float f) {
    unsigned int u = __builtin_bit_cast(unsigned int, f);
    unsigned int r = u + 0x7FFFu + ((u >> 16) & 1u);   // round-to-nearest-even
    return (unsigned short)(r >> 16);
}

// ---------------- kernel 1: W fp32 -> bf16 ----------------
__global__ void convert_w_kernel(const float* __restrict__ W,
                                 unsigned short* __restrict__ Wb) {
    size_t i = ((size_t)blockIdx.x * blockDim.x + threadIdx.x) * 4;
    float4 f = *(const float4*)(W + i);
    unsigned int lo = (unsigned int)f2bf(f.x) | ((unsigned int)f2bf(f.y) << 16);
    unsigned int hi = (unsigned int)f2bf(f.z) | ((unsigned int)f2bf(f.w) << 16);
    *(uint2*)(Wb + i) = make_uint2(lo, hi);
}

// ---------------- kernel 2: per-row ||row @ W||^4 via bf16 WMMA ----------------
// grid.x = rows / 64, block = 128 threads, dynamic LDS = 64*1032*2 bytes
__global__ void gemm_rownorm4_kernel(const float* __restrict__ X,
                                     const unsigned short* __restrict__ Wb,
                                     float* __restrict__ out_norm4) {
    extern __shared__ __align__(16) unsigned short smem[];   // [M_BLOCK][LDS_PITCH]

    const int tid  = threadIdx.x;
    const int row0 = blockIdx.x * M_BLOCK;

    // ---- stage 64x1024 fp32 panel into LDS as bf16 (coalesced float4 reads) ----
    const float4* Xv = (const float4*)(X + (size_t)row0 * DD);
    for (int i = tid; i < M_BLOCK * DD / 4; i += GEMM_THREADS) {
        float4 f = Xv[i];
        int r = i >> 8;            // D/4 = 256 float4 per row
        int c = (i & 255) << 2;
        unsigned int lo = (unsigned int)f2bf(f.x) | ((unsigned int)f2bf(f.y) << 16);
        unsigned int hi = (unsigned int)f2bf(f.z) | ((unsigned int)f2bf(f.w) << 16);
        *(uint2*)&smem[r * LDS_PITCH + c] = make_uint2(lo, hi);
    }
    __syncthreads();

    const int wave  = tid >> 5;
    const int lane  = tid & 31;
    const int m0    = wave * 16;           // 16-row strip per wave
    const int mlane = lane & 15;
    const int khalf = (lane >> 4) << 3;    // 0 for lanes 0-15, 8 for lanes 16-31

    float acc[8];
#pragma unroll
    for (int j = 0; j < 8; ++j) acc[j] = 0.0f;

    const __bf16* Wbf = (const __bf16*)Wb;
    const unsigned short* arow = &smem[(m0 + mlane) * LDS_PITCH + khalf];

    for (int nt = 0; nt < DD; nt += NT_COLS) {
        v8f c0 = {}, c1 = {}, c2 = {}, c3 = {};
        const __bf16* wcol = Wbf + nt;
        // prefetch next N-stripe's first B rows (global_prefetch_b8)
        __builtin_prefetch(wcol + NT_COLS + (size_t)lane * DD, 0, 3);
#pragma unroll 2
        for (int k0 = 0; k0 < DD; k0 += 32) {
            // ---- A fragment from LDS (documented 16-bit A 16x32 layout) ----
            const unsigned short* ap = arow + k0;
            v8bf alo = *(const v8bf*)ap;          // K = base .. base+7
            v8bf ahi = *(const v8bf*)(ap + 16);   // K = base+16 .. base+23
            v16bf a;
#pragma unroll
            for (int i = 0; i < 8; ++i) { a[i] = alo[i]; a[i + 8] = ahi[i]; }
            // ---- 4 B fragments: lane L holds W col (nt+16t+L%16), K rows in table order ----
            const __bf16* bp = wcol + (size_t)(k0 + khalf) * DD + mlane;
            v16bf b0, b1, b2, b3;
#pragma unroll
            for (int i = 0; i < 8; ++i) {
                const __bf16* r0 = bp + (size_t)i * DD;
                const __bf16* r1 = bp + (size_t)(16 + i) * DD;
                b0[i] = r0[0];  b0[i + 8] = r1[0];
                b1[i] = r0[16]; b1[i + 8] = r1[16];
                b2[i] = r0[32]; b2[i + 8] = r1[32];
                b3[i] = r0[48]; b3[i + 8] = r1[48];
            }
            c0 = __builtin_amdgcn_wmma_f32_16x16x32_bf16(false, a, false, b0, (short)0, c0, false, false);
            c1 = __builtin_amdgcn_wmma_f32_16x16x32_bf16(false, a, false, b1, (short)0, c1, false, false);
            c2 = __builtin_amdgcn_wmma_f32_16x16x32_bf16(false, a, false, b2, (short)0, c2, false, false);
            c3 = __builtin_amdgcn_wmma_f32_16x16x32_bf16(false, a, false, b3, (short)0, c3, false, false);
        }
        // square-accumulate the four 16x16 score tiles into per-lane row sums
#pragma unroll
        for (int j = 0; j < 8; ++j) {
            acc[j] = fmaf(c0[j], c0[j], acc[j]);
            acc[j] = fmaf(c1[j], c1[j], acc[j]);
            acc[j] = fmaf(c2[j], c2[j], acc[j]);
            acc[j] = fmaf(c3[j], c3[j], acc[j]);
        }
    }

    // reduce across the 16 lanes sharing a row (N dimension)
#pragma unroll
    for (int j = 0; j < 8; ++j) {
        float v = acc[j];
        v += __shfl_xor(v, 1, 32);
        v += __shfl_xor(v, 2, 32);
        v += __shfl_xor(v, 4, 32);
        v += __shfl_xor(v, 8, 32);
        acc[j] = v;
    }
    if ((lane & 15) == 0) {
        int rbase = row0 + m0 + ((lane >> 4) << 3);  // lanes>=16 hold M=j+8
#pragma unroll
        for (int j = 0; j < 8; ++j) {
            float n2 = acc[j];
            out_norm4[rbase + j] = n2 * n2;          // ||.||^4 = (||.||^2)^2
        }
    }
}

// ---------------- kernel 3: calibration (max/min over 2048 pos norms) ----------------
__global__ void calib_kernel(const float* __restrict__ pn4, float* __restrict__ ab) {
    __shared__ float smax[256], smin[256];
    int t = threadIdx.x;
    float mx = -3.4e38f, mn = 3.4e38f;
    for (int i = t; i < 2048; i += 256) {
        float v = pn4[i];
        mx = fmaxf(mx, v);
        mn = fminf(mn, v);
    }
    smax[t] = mx; smin[t] = mn;
    __syncthreads();
    for (int s = 128; s > 0; s >>= 1) {
        if (t < s) {
            smax[t] = fmaxf(smax[t], smax[t + s]);
            smin[t] = fminf(smin[t], smin[t + s]);
        }
        __syncthreads();
    }
    if (t == 0) {
        float alpha = (1.0f - 0.001f) / (smax[0] - smin[0]);
        ab[0] = alpha;
        ab[1] = 0.001f - alpha * smin[0];
    }
}

// ---------------- kernel 4: out_b = (alpha*n4_b + beta) * ||x_dot_b||^2 ----------------
// one wave per row; block = 256 threads = 8 rows
__global__ void finalize_kernel(const float* __restrict__ xdot,
                                const float* __restrict__ xn4,
                                const float* __restrict__ ab,
                                float* __restrict__ out) {
    int wave = threadIdx.x >> 5;
    int lane = threadIdx.x & 31;
    int row  = blockIdx.x * 8 + wave;
    const float4* p = (const float4*)(xdot + (size_t)row * DD);
    float s = 0.0f;
#pragma unroll
    for (int i = 0; i < 8; ++i) {
        float4 f = p[lane + i * 32];
        s = fmaf(f.x, f.x, s); s = fmaf(f.y, f.y, s);
        s = fmaf(f.z, f.z, s); s = fmaf(f.w, f.w, s);
    }
    s += __shfl_xor(s, 16, 32);
    s += __shfl_xor(s, 8, 32);
    s += __shfl_xor(s, 4, 32);
    s += __shfl_xor(s, 2, 32);
    s += __shfl_xor(s, 1, 32);
    if (lane == 0) out[row] = fmaf(ab[0], xn4[row], ab[1]) * s;
}

// ---------------- launcher ----------------
extern "C" void kernel_launch(void* const* d_in, const int* in_sizes, int n_in,
                              void* d_out, int out_size, void* d_ws, size_t ws_size,
                              hipStream_t stream) {
    const float* x    = (const float*)d_in[0];   // (32768, 1024)
    const float* xdot = (const float*)d_in[1];   // (32768, 1024)
    const float* pos  = (const float*)d_in[2];   // (2048, 1024)
    const float* W    = (const float*)d_in[3];   // (1024, 1024)
    float* out = (float*)d_out;                  // (32768,)

    const int B = 32768, P = 2048;

    // scratch layout (bytes, 256-aligned)
    char* ws = (char*)d_ws;
    unsigned short* Wb  = (unsigned short*)(ws);                       // 2 MB
    float* pos_n4       = (float*)(ws + 2 * 1024 * 1024);              // 8 KB
    float* x_n4         = (float*)(ws + 2 * 1024 * 1024 + 16384);      // 128 KB
    float* ab           = (float*)(ws + 2 * 1024 * 1024 + 16384 + 131072);

    const size_t lds_bytes = (size_t)M_BLOCK * LDS_PITCH * 2;          // 132096 B

    // 1) W -> bf16
    convert_w_kernel<<<(DD * DD) / (256 * 4), 256, 0, stream>>>(W, Wb);
    // 2) calibration GEMM-norm (pos_sample)
    gemm_rownorm4_kernel<<<P / M_BLOCK, GEMM_THREADS, lds_bytes, stream>>>(pos, Wb, pos_n4);
    // 3) alpha/beta
    calib_kernel<<<1, 256, 0, stream>>>(pos_n4, ab);
    // 4) main GEMM-norm (x)
    gemm_rownorm4_kernel<<<B / M_BLOCK, GEMM_THREADS, lds_bytes, stream>>>(x, Wb, x_n4);
    // 5) finalize with x_dot
    finalize_kernel<<<B / 8, 256, 0, stream>>>(xdot, x_n4, ab, out);
}
